// Attention_58884001628364
// MI455X (gfx1250) — compile-verified, matrix operand order checked by reading, not run on hardware
//
#include <hip/hip_runtime.h>
#include <hip/hip_bf16.h>

#define EMBED 2048
#define HEADS 16
#define BATCH 4
#define SEQ   2048
#define HD    128   // head dim
#define EE    (EMBED * EMBED)
#define BTE   (BATCH * SEQ * EMBED)

typedef __attribute__((ext_vector_type(16))) __bf16 v16bf;
typedef __attribute__((ext_vector_type(8)))  __bf16 v8bf;
typedef __attribute__((ext_vector_type(8)))  float  v8f;
typedef __attribute__((ext_vector_type(4)))  float  v4f;
typedef __attribute__((ext_vector_type(4)))  unsigned int u32x4;
typedef __attribute__((ext_vector_type(8)))  int i32x8;
typedef __attribute__((ext_vector_type(4)))  int i32x4;

static __device__ inline v8f vzero8() {
    v8f z;
#pragma unroll
    for (int i = 0; i < 8; ++i) z[i] = 0.0f;
    return z;
}

static __device__ inline v8f wmma_bf16(v16bf a, v16bf b, v8f c) {
    // D = A(16x32 bf16) * B(32x16 bf16) + C(16x16 f32)
    return __builtin_amdgcn_wmma_f32_16x16x32_bf16(
        false, a, false, b, (short)0, c, false, false);
}

// A-matrix (16x32, row-major, contiguous along K) fragment, bf16 source.
// Also serves as the B-matrix when B is stored B^T row-major (rows = N,
// contiguous along K): weights, keys, transposed V, LDS tiles.
static __device__ inline v16bf ld_frag_bf16(const __bf16* p, int ld) {
    int lane = threadIdx.x & 31;
    int row  = lane & 15;
    int kb   = (lane >> 4) << 3;          // 0 or 8
    const __bf16* r = p + (size_t)row * ld + kb;
    v8bf lo = *(const v8bf*)(r);          // K = kb .. kb+7
    v8bf hi = *(const v8bf*)(r + 16);     // K = 16+kb .. 16+kb+7
    v16bf out;
#pragma unroll
    for (int i = 0; i < 8; ++i) { out[i] = lo[i]; out[i + 8] = hi[i]; }
    return out;
}

// Same layout, f32 source converted to bf16 in-register.
static __device__ inline v16bf ld_frag_f32(const float* p, int ld) {
    int lane = threadIdx.x & 31;
    int row  = lane & 15;
    int kb   = (lane >> 4) << 3;
    const float* r = p + (size_t)row * ld + kb;
    v4f a0 = *(const v4f*)(r);
    v4f a1 = *(const v4f*)(r + 4);
    v4f b0 = *(const v4f*)(r + 16);
    v4f b1 = *(const v4f*)(r + 20);
    v16bf out;
#pragma unroll
    for (int i = 0; i < 4; ++i) {
        out[i]      = (__bf16)a0[i];
        out[i + 4]  = (__bf16)a1[i];
        out[i + 8]  = (__bf16)b0[i];
        out[i + 12] = (__bf16)b1[i];
    }
    return out;
}

// ---- TDM: async-DMA a 2D bf16 tile (tile_x contiguous elems, tile_y rows,
// row stride stride_x elems) from global memory into LDS at lds_off. ----
static __device__ inline void tdm_load_2d_bf16(unsigned int lds_off,
                                               const void* gptr,
                                               int tile_x, int tile_y,
                                               int stride_x) {
    unsigned long long ga = (unsigned long long)gptr;
    u32x4 g0;
    g0[0] = 1u;                                   // count=1, user descriptor
    g0[1] = lds_off;                              // LDS byte address
    g0[2] = (unsigned int)ga;                     // global_addr[31:0]
    g0[3] = (unsigned int)((ga >> 32) & 0x1FFFFFFu) | (2u << 30); // addr[56:32] | type=2
    i32x8 g1;
    g1[0] = (1 << 16);                            // data_size = 1 (2 bytes)
    g1[1] = (int)((unsigned)tile_x << 16);        // tensor_dim0[15:0] (== tile_x)
    g1[2] = (int)(((unsigned)tile_x >> 16) | ((unsigned)tile_y << 16)); // dim0 hi | dim1 lo
    g1[3] = (int)((((unsigned)tile_y >> 16) & 0xffffu) | ((unsigned)tile_x << 16)); // dim1 hi | tile_dim0
    g1[4] = (int)((unsigned)tile_y & 0xffffu);    // tile_dim1 (tile_dim2 = 0)
    g1[5] = stride_x;                             // tensor_dim0_stride[31:0]
    g1[6] = 0;
    g1[7] = 0;
    i32x4 gz4 = {0, 0, 0, 0};
    i32x8 gz8 = {0, 0, 0, 0, 0, 0, 0, 0};
    // clang-23 / therock-10.0 toolchain: 6-arg form
    __builtin_amdgcn_tensor_load_to_lds(g0, g1, gz4, gz4, gz8, 0);
}

// ---------------- weight f32 -> bf16 convert ----------------
__global__ void cvt_bf16_kernel(const float* __restrict__ src,
                                __bf16* __restrict__ dst, int n) {
    int stride = gridDim.x * blockDim.x;
    for (int i = blockIdx.x * blockDim.x + threadIdx.x; i < n; i += stride)
        dst[i] = (__bf16)src[i];
}

// ---------------- fused QKV projection -------------------
// z=0: Q' -> Qp [B,H,T,HD]; z=1: K' -> Kp [B,H,T,HD]; z=2: V' -> Vt [B,H,HD,T]
__global__ __launch_bounds__(256) void qkv_gemm_kernel(
    const float* __restrict__ qin, const float* __restrict__ kin,
    const float* __restrict__ vin,
    const __bf16* __restrict__ wq, const __bf16* __restrict__ wk,
    const __bf16* __restrict__ wv,
    __bf16* __restrict__ Qp, __bf16* __restrict__ Kp, __bf16* __restrict__ Vt) {
    int z = blockIdx.z;
    const float*  A = (z == 0) ? qin : (z == 1) ? kin : vin;
    const __bf16* W = (z == 0) ? wq  : (z == 1) ? wk  : wv;

    int wave = threadIdx.x >> 5;
    int lane = threadIdx.x & 31;
    int half = lane >> 4;
    int col  = lane & 15;
    int mt   = blockIdx.x;                       // 32-row tile, 0..255
    int n0   = blockIdx.y * 512 + wave * 64;     // this wave's 64 columns

    const float* Ab = A + (size_t)mt * 32 * EMBED;

    v8f acc[8];                                  // [ms*4 + nt]
#pragma unroll
    for (int i = 0; i < 8; ++i) acc[i] = vzero8();

    for (int kk = 0; kk < EMBED; kk += 32) {
        __builtin_prefetch(Ab + kk + 128, 0, 3);
        v16bf a0 = ld_frag_f32(Ab + kk, EMBED);
        v16bf a1 = ld_frag_f32(Ab + (size_t)16 * EMBED + kk, EMBED);
#pragma unroll
        for (int nt = 0; nt < 4; ++nt) {
            v16bf b = ld_frag_bf16(W + (size_t)(n0 + nt * 16) * EMBED + kk, EMBED);
            acc[nt]     = wmma_bf16(a0, b, acc[nt]);
            acc[4 + nt] = wmma_bf16(a1, b, acc[4 + nt]);
        }
    }

#pragma unroll
    for (int ms = 0; ms < 2; ++ms) {
#pragma unroll
        for (int nt = 0; nt < 4; ++nt) {
#pragma unroll
            for (int r = 0; r < 8; ++r) {
                int m = mt * 32 + ms * 16 + r + 8 * half;   // b*SEQ + t
                int n = n0 + nt * 16 + col;                 // h*HD + d
                int bb = m >> 11, t = m & (SEQ - 1);
                int hh = n >> 7,  d = n & (HD - 1);
                __bf16 val = (__bf16)acc[ms * 4 + nt][r];
                if (z == 2)
                    Vt[(((size_t)(bb * HEADS + hh)) * HD + d) * SEQ + t] = val;
                else if (z == 0)
                    Qp[(((size_t)(bb * HEADS + hh)) * SEQ + t) * HD + d] = val;
                else
                    Kp[(((size_t)(bb * HEADS + hh)) * SEQ + t) * HD + d] = val;
            }
        }
    }
}

// ---------------- flash attention core -------------------
// 1 wave per (b, h, 16-row q tile). Online softmax; V tiles DMA'd by the
// Tensor Data Mover into double-buffered LDS; 1KB LDS stage for the P-tile
// C-layout -> A-layout transpose.
__global__ __launch_bounds__(32) void attn_kernel(
    const __bf16* __restrict__ Qp, const __bf16* __restrict__ Kp,
    const __bf16* __restrict__ Vt, __bf16* __restrict__ attn_out) {
    extern __shared__ char smem[];
    __bf16* stage = (__bf16*)smem;                    // [16][32] P-tile
    // V tile double buffers live at smem+1024 and smem+1024+8192 ([128][32] each)

    int qt   = blockIdx.x;
    int h    = blockIdx.y;
    int b    = blockIdx.z;
    int lane = threadIdx.x;
    int half = lane >> 4;
    int col  = lane & 15;

    const __bf16* qph = Qp + ((size_t)(b * HEADS + h)) * SEQ * HD;
    const __bf16* kph = Kp + ((size_t)(b * HEADS + h)) * SEQ * HD;
    const __bf16* vth = Vt + ((size_t)(b * HEADS + h)) * HD * SEQ;

    v16bf aq[4];
#pragma unroll
    for (int kd = 0; kd < 4; ++kd)
        aq[kd] = ld_frag_bf16(qph + (size_t)qt * 16 * HD + kd * 32, HD);

    const float scale = 0.08838834764831845f;  // 1/sqrt(128)
    float m[8], l[8];
#pragma unroll
    for (int r = 0; r < 8; ++r) { m[r] = -3.0e38f; l[r] = 0.0f; }
    v8f accs[8];
#pragma unroll
    for (int dt = 0; dt < 8; ++dt) accs[dt] = vzero8();

    // prefetch first V tile: 32 keys x 128 dims
    tdm_load_2d_bf16(1024u, vth, 32, HD, SEQ);

    for (int jk = 0; jk < SEQ / 32; ++jk) {
        if (jk + 1 < SEQ / 32)
            tdm_load_2d_bf16(1024u + (unsigned)(((jk + 1) & 1) * 8192),
                             vth + (jk + 1) * 32, 32, HD, SEQ);

        // ---- scores for 32 keys: two 16-col tiles ----
        v8f s0 = vzero8(), s1 = vzero8();
#pragma unroll
        for (int kd = 0; kd < 4; ++kd) {
            v16bf bk0 = ld_frag_bf16(kph + (size_t)(jk * 32) * HD + kd * 32, HD);
            v16bf bk1 = ld_frag_bf16(kph + (size_t)(jk * 32 + 16) * HD + kd * 32, HD);
            s0 = wmma_bf16(aq[kd], bk0, s0);
            s1 = wmma_bf16(aq[kd], bk1, s1);
        }

        // ---- online softmax update ----
        float p0[8], p1[8], factor[8], red[8];
#pragma unroll
        for (int r = 0; r < 8; ++r) {
            p0[r] = s0[r] * scale;
            p1[r] = s1[r] * scale;
            red[r] = fmaxf(p0[r], p1[r]);
        }
#pragma unroll
        for (int mask = 1; mask <= 8; mask <<= 1)
#pragma unroll
            for (int r = 0; r < 8; ++r)
                red[r] = fmaxf(red[r], __shfl_xor(red[r], mask, 32));
#pragma unroll
        for (int r = 0; r < 8; ++r) {
            float mn = fmaxf(m[r], red[r]);
            factor[r] = __expf(m[r] - mn);
            m[r] = mn;
            p0[r] = __expf(p0[r] - mn);
            p1[r] = __expf(p1[r] - mn);
            red[r] = p0[r] + p1[r];
        }
#pragma unroll
        for (int mask = 1; mask <= 8; mask <<= 1)
#pragma unroll
            for (int r = 0; r < 8; ++r)
                red[r] += __shfl_xor(red[r], mask, 32);
#pragma unroll
        for (int r = 0; r < 8; ++r) l[r] = l[r] * factor[r] + red[r];

        // rescale running output accumulators
#pragma unroll
        for (int dt = 0; dt < 8; ++dt)
#pragma unroll
            for (int r = 0; r < 8; ++r) accs[dt][r] *= factor[r];

        // stage P (C-layout) to LDS as row-major [16][32] for A-frag reload
#pragma unroll
        for (int r = 0; r < 8; ++r) {
            stage[(r + 8 * half) * 32 + col]      = (__bf16)p0[r];
            stage[(r + 8 * half) * 32 + 16 + col] = (__bf16)p1[r];
        }
        v16bf ap = ld_frag_bf16(stage, 32);

        // ---- P @ V: wait for this iteration's V tile (keep next in flight) ----
        if (jk + 1 < SEQ / 32)
            __builtin_amdgcn_s_wait_tensorcnt(1);
        else
            __builtin_amdgcn_s_wait_tensorcnt(0);

        const __bf16* vb = (const __bf16*)(smem + 1024 + (jk & 1) * 8192);
#pragma unroll
        for (int dt = 0; dt < 8; ++dt) {
            v16bf bv = ld_frag_bf16(vb + dt * 16 * 32, 32);
            accs[dt] = wmma_bf16(ap, bv, accs[dt]);
        }
    }

    float inv[8];
#pragma unroll
    for (int r = 0; r < 8; ++r) inv[r] = 1.0f / l[r];

#pragma unroll
    for (int dt = 0; dt < 8; ++dt) {
#pragma unroll
        for (int r = 0; r < 8; ++r) {
            int trow = qt * 16 + r + 8 * half;
            int n    = h * HD + dt * 16 + col;
            attn_out[((size_t)(b * SEQ + trow)) * EMBED + n] =
                (__bf16)(accs[dt][r] * inv[r]);
        }
    }
}

// ---------------- output projection + bias ----------------
__global__ __launch_bounds__(256) void out_gemm_kernel(
    const __bf16* __restrict__ X, const __bf16* __restrict__ wo,
    const float* __restrict__ bo, float* __restrict__ out) {
    int wave = threadIdx.x >> 5;
    int lane = threadIdx.x & 31;
    int half = lane >> 4;
    int col  = lane & 15;
    int mt   = blockIdx.x;                    // 32-row tile
    int n0   = blockIdx.y * 512 + wave * 64;

    const __bf16* Ab = X + (size_t)mt * 32 * EMBED;

    v8f acc[8];
#pragma unroll
    for (int i = 0; i < 8; ++i) acc[i] = vzero8();

    for (int kk = 0; kk < EMBED; kk += 32) {
        __builtin_prefetch(Ab + kk + 128, 0, 3);
        v16bf a0 = ld_frag_bf16(Ab + kk, EMBED);
        v16bf a1 = ld_frag_bf16(Ab + (size_t)16 * EMBED + kk, EMBED);
#pragma unroll
        for (int nt = 0; nt < 4; ++nt) {
            v16bf b = ld_frag_bf16(wo + (size_t)(n0 + nt * 16) * EMBED + kk, EMBED);
            acc[nt]     = wmma_bf16(a0, b, acc[nt]);
            acc[4 + nt] = wmma_bf16(a1, b, acc[4 + nt]);
        }
    }

#pragma unroll
    for (int ms = 0; ms < 2; ++ms) {
#pragma unroll
        for (int nt = 0; nt < 4; ++nt) {
            int n = n0 + nt * 16 + col;
            float bias = bo[n];
#pragma unroll
            for (int r = 0; r < 8; ++r) {
                int mrow = mt * 32 + ms * 16 + r + 8 * half;
                out[(size_t)mrow * EMBED + n] = acc[ms * 4 + nt][r] + bias;
            }
        }
    }
}

extern "C" void kernel_launch(void* const* d_in, const int* in_sizes, int n_in,
                              void* d_out, int out_size, void* d_ws, size_t ws_size,
                              hipStream_t stream) {
    // setup_inputs() order: k, q, v, Wk, Wq, Wv, Wo, bo
    const float* kin = (const float*)d_in[0];
    const float* qin = (const float*)d_in[1];
    const float* vin = (const float*)d_in[2];
    const float* Wk  = (const float*)d_in[3];
    const float* Wq  = (const float*)d_in[4];
    const float* Wv  = (const float*)d_in[5];
    const float* Wo  = (const float*)d_in[6];
    const float* bo  = (const float*)d_in[7];
    float* out = (float*)d_out;

    char* w = (char*)d_ws;
    __bf16* wqb = (__bf16*)w; w += (size_t)EE * 2;
    __bf16* wkb = (__bf16*)w; w += (size_t)EE * 2;
    __bf16* wvb = (__bf16*)w; w += (size_t)EE * 2;
    __bf16* wob = (__bf16*)w; w += (size_t)EE * 2;
    __bf16* Qp  = (__bf16*)w; w += (size_t)BTE * 2;
    __bf16* Kp  = (__bf16*)w; w += (size_t)BTE * 2;
    __bf16* Vt  = (__bf16*)w; w += (size_t)BTE * 2;
    __bf16* Xo  = (__bf16*)w; w += (size_t)BTE * 2;

    cvt_bf16_kernel<<<2048, 256, 0, stream>>>(Wq, wqb, EE);
    cvt_bf16_kernel<<<2048, 256, 0, stream>>>(Wk, wkb, EE);
    cvt_bf16_kernel<<<2048, 256, 0, stream>>>(Wv, wvb, EE);
    cvt_bf16_kernel<<<2048, 256, 0, stream>>>(Wo, wob, EE);

    dim3 ggrid(BATCH * SEQ / 32, EMBED / 512, 3);
    qkv_gemm_kernel<<<ggrid, 256, 0, stream>>>(qin, kin, vin, wqb, wkb, wvb,
                                               Qp, Kp, Vt);

    dim3 agrid(SEQ / 16, HEADS, BATCH);
    size_t lds = 1024 + 2 * 8192;   // P stage + double-buffered V tiles
    attn_kernel<<<agrid, 32, lds, stream>>>(Qp, Kp, Vt, Xo);

    dim3 ogrid(BATCH * SEQ / 32, EMBED / 512, 1);
    out_gemm_kernel<<<ogrid, 256, 0, stream>>>(Xo, wob, bo, out);
}
